// MINICPM_22333829940008
// MI455X (gfx1250) — matrix operand order repaired and checked by reading, not compiled, not run on hardware
//
#include <hip/hip_runtime.h>
#include <hip/hip_bf16.h>
#include <math.h>

// ---- problem constants (match reference) ----
#define L_    2
#define D_    2048
#define H_    32
#define KVH_  8
#define HD_   64
#define FF_   8192
#define V_    32768
#define S_    1024
#define HIST_ 1024
#define T_    (HIST_ + S_)
#define EPSF  1e-6f

typedef __bf16 v16bf __attribute__((ext_vector_type(16)));
typedef float  v8f   __attribute__((ext_vector_type(8)));

#if defined(__has_builtin)
#if __has_builtin(__builtin_amdgcn_tensor_load_to_lds) && \
    __has_builtin(__builtin_amdgcn_tensor_store_from_lds) && \
    __has_builtin(__builtin_amdgcn_s_wait_tensorcnt)
#define HAVE_TDM 1
#endif
#if __has_builtin(__builtin_amdgcn_cvt_pk_bf16_f32)
#define HAVE_PK_CVT 1
#endif
#endif

// fp32 -> bf16, round-half-up (1 add + shift)
__device__ inline unsigned short f2bf(float f) {
  unsigned int u = __builtin_bit_cast(unsigned int, f);
  return (unsigned short)((u + 0x8000u) >> 16);
}
// two floats -> packed bf16x2 in one dword (elem0 in low 16)
__device__ inline unsigned int pk_bf16(float x, float y) {
#if defined(HAVE_PK_CVT)
  auto r = __builtin_amdgcn_cvt_pk_bf16_f32(x, y);
  return __builtin_bit_cast(unsigned int, r);
#else
  unsigned int ux = __builtin_bit_cast(unsigned int, x) + 0x8000u;
  unsigned int uy = __builtin_bit_cast(unsigned int, y) + 0x8000u;
  return __builtin_amdgcn_perm(uy, ux, 0x07060302u);   // {uy.hi, ux.hi}
#endif
}
__device__ inline v8f zero8() {
  v8f z;
#pragma unroll
  for (int r = 0; r < 8; ++r) z[r] = 0.0f;
  return z;
}
__device__ inline void pack4(unsigned short* dst, float x0, float x1, float x2, float x3) {
  uint2 u;
  u.x = pk_bf16(x0, x1);
  u.y = pk_bf16(x2, x3);
  *(uint2*)dst = u;
}
__device__ inline v16bf frag16(float4 a0, float4 a1, float4 a2, float4 a3) {
  union { unsigned int w[8]; v16bf v; } u;
  u.w[0] = pk_bf16(a0.x, a0.y); u.w[1] = pk_bf16(a0.z, a0.w);
  u.w[2] = pk_bf16(a1.x, a1.y); u.w[3] = pk_bf16(a1.z, a1.w);
  u.w[4] = pk_bf16(a2.x, a2.y); u.w[5] = pk_bf16(a2.z, a2.w);
  u.w[6] = pk_bf16(a3.x, a3.y); u.w[7] = pk_bf16(a3.z, a3.w);
  return u.v;
}

// ---- swizzled-layout index helpers ----
// A-matrix (WMMA A fragment layout), matrix [M x K] bf16:
__device__ inline size_t aswz_idx(int row, int k, int K) {
  int mt = row >> 4, sb = k >> 5, ki = k & 31;
  int ha = (ki >> 3) & 1;
  int e  = (ki & 7) + ((ki >> 4) << 3);
  int lane = ha * 16 + (row & 15);
  return (((size_t)mt * (K >> 5) + sb) * 32 + lane) * 16 + e;
}

// =====================================================================
// Weight swizzle: fp32 [K x N] -> bf16 in WMMA B-fragment order.
// =====================================================================
__global__ __launch_bounds__(256) void wswz_kernel(
    const float* __restrict__ B, unsigned short* __restrict__ Bs, int N, int K)
{
  size_t idx = (size_t)blockIdx.x * 256 + threadIdx.x;
  if (idx >= (size_t)K * N / 4) return;
  int n = (int)(idx % N);
  int k = (int)(idx / N) * 4;
  const float* bp = B + (size_t)k * N + n;
  float x0 = bp[0];
  float x1 = bp[(size_t)N];
  float x2 = bp[(size_t)2 * N];
  float x3 = bp[(size_t)3 * N];
  int sb = k >> 5, ki = k & 31, hb = (ki >> 3) & 1, e0 = (ki & 7) + ((ki >> 4) << 3), nt = n >> 4;
  size_t d = (((size_t)sb * (N >> 4) + nt) * 32 + (hb * 16 + (n & 15))) * 16 + e0;
  pack4(&Bs[d], x0, x1, x2, x3);
}

// =====================================================================
// GEMM on pre-swizzled bf16 operands: C[M,N] (+)= A*B, fp32 C.
// Block tile 128x128x32, 8 waves, wave tile 32x64.
// A fragments: direct global 32B loads, software-pipelined one step ahead.
// B tile: contiguous 8KB/k-step, async global->LDS, double buffered.
// =====================================================================
#define BM 128
#define BN 128
#define BK 32

__device__ inline void async_cp32(const unsigned short* gsrc, unsigned short* ldst) {
  unsigned lo = (unsigned int)(unsigned long long)(uintptr_t)ldst;
  unsigned long long ga = (unsigned long long)(uintptr_t)gsrc;
  asm volatile("global_load_async_to_lds_b128 %0, %1, off" :: "v"(lo), "v"(ga) : "memory");
  unsigned lo2 = lo + 16;
  unsigned long long ga2 = ga + 16;
  asm volatile("global_load_async_to_lds_b128 %0, %1, off" :: "v"(lo2), "v"(ga2) : "memory");
}

__global__ __launch_bounds__(256) void gemm_sw(
    const unsigned short* __restrict__ As,   // swizzled A  [M x K]
    const unsigned short* __restrict__ Bs,   // swizzled B  [K x N]
    float* __restrict__ C, int M, int N, int K, int accum)
{
  __shared__ __align__(32) unsigned short Bt[2][4096];   // 2 x 8KB

  const int tid  = threadIdx.x;
  const int lane = tid & 31;
  const int wave = tid >> 5;
  const int wm   = wave >> 1;               // 0..3
  const int wn   = wave & 1;                // 0..1
  const int m0   = blockIdx.y * BM;
  const int n0   = blockIdx.x * BN;
  const int half = lane >> 4;
  const int lr   = lane & 15;

  const int nk = K >> 5;                    // >= 2 always here
  const unsigned short* a0 =
      As + (((size_t)((m0 >> 4) + wm * 2 + 0) * nk) * 32 + lane) * 16;
  const unsigned short* a1 =
      As + (((size_t)((m0 >> 4) + wm * 2 + 1) * nk) * 32 + lane) * 16;
  const size_t bstep = (size_t)(N >> 4) * 512;
  const unsigned short* bsrc = Bs + (size_t)(n0 >> 4) * 512 + tid * 16;

  v8f acc[2][4];
#pragma unroll
  for (int i = 0; i < 2; ++i)
#pragma unroll
    for (int j = 0; j < 4; ++j) acc[i][j] = zero8();

  // prologue: stage 0 B in flight; stage 0 A fragments in registers
  async_cp32(bsrc, &Bt[0][tid * 16]);
  const unsigned short* bnext = bsrc + bstep;
  v16bf af0 = *(const v16bf*)(a0);
  v16bf af1 = *(const v16bf*)(a1);
  a0 += 512; a1 += 512;

  for (int s = 0; s < nk - 1; ++s) {
    const int cur = s & 1;
    async_cp32(bnext, &Bt[cur ^ 1][tid * 16]);     // overlap next B stage
    bnext += bstep;
    asm volatile("s_wait_asynccnt 2" ::: "memory"); // this stage's 2 done
    __syncthreads();

    const v16bf* bt = (const v16bf*)&Bt[cur][0];
    v16bf b0 = bt[(wn * 4 + 0) * 32 + lane];
    v16bf b1 = bt[(wn * 4 + 1) * 32 + lane];
    v16bf b2 = bt[(wn * 4 + 2) * 32 + lane];
    v16bf b3 = bt[(wn * 4 + 3) * 32 + lane];

    // prefetch next A fragments; latency hides under the WMMA chain
    v16bf an0 = *(const v16bf*)(a0);
    v16bf an1 = *(const v16bf*)(a1);
    a0 += 512; a1 += 512;

    acc[0][0] = __builtin_amdgcn_wmma_f32_16x16x32_bf16(false, af0, false, b0, (short)0, acc[0][0], false, false);
    acc[1][0] = __builtin_amdgcn_wmma_f32_16x16x32_bf16(false, af1, false, b0, (short)0, acc[1][0], false, false);
    acc[0][1] = __builtin_amdgcn_wmma_f32_16x16x32_bf16(false, af0, false, b1, (short)0, acc[0][1], false, false);
    acc[1][1] = __builtin_amdgcn_wmma_f32_16x16x32_bf16(false, af1, false, b1, (short)0, acc[1][1], false, false);
    acc[0][2] = __builtin_amdgcn_wmma_f32_16x16x32_bf16(false, af0, false, b2, (short)0, acc[0][2], false, false);
    acc[1][2] = __builtin_amdgcn_wmma_f32_16x16x32_bf16(false, af1, false, b2, (short)0, acc[1][2], false, false);
    acc[0][3] = __builtin_amdgcn_wmma_f32_16x16x32_bf16(false, af0, false, b3, (short)0, acc[0][3], false, false);
    acc[1][3] = __builtin_amdgcn_wmma_f32_16x16x32_bf16(false, af1, false, b3, (short)0, acc[1][3], false, false);

    af0 = an0; af1 = an1;
    __syncthreads();
  }

  // final k-step (no further B stage in flight)
  asm volatile("s_wait_asynccnt 0" ::: "memory");
  __syncthreads();
  {
    const v16bf* bt = (const v16bf*)&Bt[(nk - 1) & 1][0];
    v16bf b0 = bt[(wn * 4 + 0) * 32 + lane];
    v16bf b1 = bt[(wn * 4 + 1) * 32 + lane];
    v16bf b2 = bt[(wn * 4 + 2) * 32 + lane];
    v16bf b3 = bt[(wn * 4 + 3) * 32 + lane];
    acc[0][0] = __builtin_amdgcn_wmma_f32_16x16x32_bf16(false, af0, false, b0, (short)0, acc[0][0], false, false);
    acc[1][0] = __builtin_amdgcn_wmma_f32_16x16x32_bf16(false, af1, false, b0, (short)0, acc[1][0], false, false);
    acc[0][1] = __builtin_amdgcn_wmma_f32_16x16x32_bf16(false, af0, false, b1, (short)0, acc[0][1], false, false);
    acc[1][1] = __builtin_amdgcn_wmma_f32_16x16x32_bf16(false, af1, false, b1, (short)0, acc[1][1], false, false);
    acc[0][2] = __builtin_amdgcn_wmma_f32_16x16x32_bf16(false, af0, false, b2, (short)0, acc[0][2], false, false);
    acc[1][2] = __builtin_amdgcn_wmma_f32_16x16x32_bf16(false, af1, false, b2, (short)0, acc[1][2], false, false);
    acc[0][3] = __builtin_amdgcn_wmma_f32_16x16x32_bf16(false, af0, false, b3, (short)0, acc[0][3], false, false);
    acc[1][3] = __builtin_amdgcn_wmma_f32_16x16x32_bf16(false, af1, false, b3, (short)0, acc[1][3], false, false);
  }

#pragma unroll
  for (int i = 0; i < 2; ++i)
#pragma unroll
    for (int j = 0; j < 4; ++j)
#pragma unroll
      for (int r = 0; r < 8; ++r) {
        int row = m0 + wm * 32 + i * 16 + r + (half << 3);
        int col = n0 + wn * 64 + j * 16 + lr;
        size_t o = (size_t)row * N + col;
        float v = acc[i][j][r];
        C[o] = accum ? (C[o] + v) : v;
      }
}

// =====================================================================
// RMSNorm -> swizzled bf16 A-layout (feeds GEMMs). One block per row.
// =====================================================================
__global__ __launch_bounds__(256) void rmsnorm_sw_kernel(
    const float* __restrict__ x, const float* __restrict__ w,
    unsigned short* __restrict__ ys, int n)
{
  __shared__ float red[8];
  int row = blockIdx.x;
  const float* xr = x + (size_t)row * n;
  float s = 0.f;
  for (int i = threadIdx.x; i < n; i += 256) { float v = xr[i]; s += v * v; }
#pragma unroll
  for (int m = 16; m >= 1; m >>= 1) s += __shfl_xor(s, m, 32);
  if ((threadIdx.x & 31) == 0) red[threadIdx.x >> 5] = s;
  __syncthreads();
  if (threadIdx.x < 8) {
    float t = red[threadIdx.x];
#pragma unroll
    for (int m = 4; m >= 1; m >>= 1) t += __shfl_xor(t, m, 32);
    if (threadIdx.x == 0) red[0] = t;
  }
  __syncthreads();
  float inv = rsqrtf(red[0] / (float)n + EPSF);
  for (int i = threadIdx.x * 4; i < n; i += 1024) {
    float4 v  = *(const float4*)(xr + i);
    float4 wv = *(const float4*)(w + i);
    pack4(&ys[aswz_idx(row, i, n)],
          v.x * inv * wv.x, v.y * inv * wv.y, v.z * inv * wv.z, v.w * inv * wv.w);
  }
}

// plain fp32 RMSNorm (final norm before lm_head)
__global__ __launch_bounds__(256) void rmsnorm_kernel(
    const float* __restrict__ x, const float* __restrict__ w,
    float* __restrict__ y, int n)
{
  __shared__ float red[8];
  const float* xr = x;
  float s = 0.f;
  for (int i = threadIdx.x; i < n; i += 256) { float v = xr[i]; s += v * v; }
#pragma unroll
  for (int m = 16; m >= 1; m >>= 1) s += __shfl_xor(s, m, 32);
  if ((threadIdx.x & 31) == 0) red[threadIdx.x >> 5] = s;
  __syncthreads();
  if (threadIdx.x < 8) {
    float t = red[threadIdx.x];
#pragma unroll
    for (int m = 4; m >= 1; m >>= 1) t += __shfl_xor(t, m, 32);
    if (threadIdx.x == 0) red[0] = t;
  }
  __syncthreads();
  float inv = rsqrtf(red[0] / (float)n + EPSF);
  for (int i = threadIdx.x; i < n; i += 256) y[i] = xr[i] * inv * w[i];
}

// =====================================================================
// RoPE on q in-place.
// =====================================================================
__global__ void rope_q_kernel(float* __restrict__ q,
                              const float* __restrict__ cosT,
                              const float* __restrict__ sinT)
{
  int idx = blockIdx.x * blockDim.x + threadIdx.x;
  if (idx >= S_ * H_ * (HD_ / 2)) return;
  int d = idx & 31;
  int h = (idx >> 5) & (H_ - 1);
  int s = idx >> 10;
  int pos = HIST_ + s;
  size_t base = (size_t)s * (H_ * HD_) + h * HD_;
  float x1 = q[base + d], x2 = q[base + d + 32];
  q[base + d]      = x1 * cosT[pos * HD_ + d]      - x2 * sinT[pos * HD_ + d];
  q[base + d + 32] = x2 * cosT[pos * HD_ + d + 32] + x1 * sinT[pos * HD_ + d + 32];
}

// =====================================================================
// k (RoPE) and v scattered into the [KVH,T,HD] caches for one layer.
// =====================================================================
__global__ void kv_cache_kernel(const float* __restrict__ kb,
                                const float* __restrict__ vb,
                                float* __restrict__ ok, float* __restrict__ ov,
                                const float* __restrict__ cosT,
                                const float* __restrict__ sinT)
{
  int idx = blockIdx.x * blockDim.x + threadIdx.x;
  if (idx >= S_ * KVH_ * HD_) return;
  int d  = idx & 63;
  int kv = (idx >> 6) & (KVH_ - 1);
  int s  = idx >> 9;
  size_t src = (size_t)s * (KVH_ * HD_) + kv * HD_ + d;
  size_t dst = ((size_t)kv * T_ + (HIST_ + s)) * HD_ + d;
  ov[dst] = vb[src];
  if (d < 32) {
    int pos = HIST_ + s;
    float x1 = kb[src], x2 = kb[src + 32];
    ok[dst]      = x1 * cosT[pos * HD_ + d]      - x2 * sinT[pos * HD_ + d];
    ok[dst + 32] = x2 * cosT[pos * HD_ + d + 32] + x1 * sinT[pos * HD_ + d + 32];
  }
}

#if defined(HAVE_TDM)
typedef unsigned int v4u_ __attribute__((ext_vector_type(4)));
typedef int          v8i_ __attribute__((ext_vector_type(8)));
typedef int          v4i_ __attribute__((ext_vector_type(4)));

__device__ inline void tdm_copy_2d(unsigned int lds_off, const void* gaddr,
                                   unsigned dim0, unsigned dim1, unsigned stride0,
                                   bool store)
{
  unsigned long long ga = (unsigned long long)(uintptr_t)gaddr;
  v4u_ g0;
  g0[0] = 1u;
  g0[1] = lds_off;
  g0[2] = (unsigned)(ga & 0xFFFFFFFFu);
  g0[3] = (unsigned)((ga >> 32) & 0x1FFFFFFu) | 0x80000000u;
  v8i_ g1;
  g1[0] = 0x00020000;
  g1[1] = (int)(dim0 << 16);
  g1[2] = (int)((dim0 >> 16) | (dim1 << 16));
  g1[3] = (int)((dim1 >> 16) | (dim0 << 16));
  g1[4] = (int)dim1;
  g1[5] = (int)stride0;
  g1[6] = 0;
  g1[7] = 0;
  v4i_ gz4; gz4[0] = 0; gz4[1] = 0; gz4[2] = 0; gz4[3] = 0;
  v8i_ gz8;
#pragma unroll
  for (int i = 0; i < 8; ++i) gz8[i] = 0;
  if (store) __builtin_amdgcn_tensor_store_from_lds(g0, g1, gz4, gz4, gz8, 0);
  else       __builtin_amdgcn_tensor_load_to_lds(g0, g1, gz4, gz4, gz8, 0);
}
#endif

// =====================================================================
// Past K/V -> caches. K via TDM (512x32 fp32 tile), V via async LDS.
// grid = (4, L*KVH).
// =====================================================================
__global__ __launch_bounds__(256) void past_copy_kernel(
    const float* __restrict__ pk, const float* __restrict__ pv,
    float* __restrict__ ok, float* __restrict__ ov)
{
  __shared__ float stg_k[16384];
  __shared__ float stg_v[4096];
  const int lk  = blockIdx.y;
  const size_t srcb = (size_t)lk * HIST_ * HD_ + (size_t)blockIdx.x * 16384;
  const size_t dstb = (size_t)lk * T_ * HD_ + (size_t)blockIdx.x * 16384;

#if defined(HAVE_TDM)
  if (threadIdx.x < 32) {
    unsigned int lds0 = (unsigned int)(unsigned long long)(uintptr_t)&stg_k[0];
    tdm_copy_2d(lds0, pk + srcb, 512u, 32u, 512u, false);
    __builtin_amdgcn_s_wait_tensorcnt(0);
    tdm_copy_2d(lds0, ok + dstb, 512u, 32u, 512u, true);
    __builtin_amdgcn_s_wait_tensorcnt(0);
  }
#else
  for (int p = 0; p < 4; ++p) {
    size_t base = (size_t)p * 4096;
#pragma unroll
    for (int i = 0; i < 4; ++i) {
      unsigned off = threadIdx.x * 4 + i * 1024;
      unsigned lo = (unsigned int)(unsigned long long)(uintptr_t)&stg_k[off];
      unsigned long long gaL = (unsigned long long)(uintptr_t)(pk + srcb + base + off);
      asm volatile("global_load_async_to_lds_b128 %0, %1, off" :: "v"(lo), "v"(gaL) : "memory");
    }
    asm volatile("s_wait_asynccnt 0" ::: "memory");
#pragma unroll
    for (int i = 0; i < 4; ++i) {
      unsigned off = threadIdx.x * 4 + i * 1024;
      unsigned lo = (unsigned int)(unsigned long long)(uintptr_t)&stg_k[off];
      unsigned long long gaS = (unsigned long long)(uintptr_t)(ok + dstb + base + off);
      asm volatile("global_store_async_from_lds_b128 %0, %1, off" :: "v"(gaS), "v"(lo) : "memory");
    }
    asm volatile("s_wait_asynccnt 0" ::: "memory");
  }
#endif

  for (int p = 0; p < 4; ++p) {
    size_t base = (size_t)p * 4096;
#pragma unroll
    for (int i = 0; i < 4; ++i) {
      unsigned off = threadIdx.x * 4 + i * 1024;
      unsigned lo = (unsigned int)(unsigned long long)(uintptr_t)&stg_v[off];
      unsigned long long gaL = (unsigned long long)(uintptr_t)(pv + srcb + base + off);
      asm volatile("global_load_async_to_lds_b128 %0, %1, off" :: "v"(lo), "v"(gaL) : "memory");
    }
    asm volatile("s_wait_asynccnt 0" ::: "memory");
#pragma unroll
    for (int i = 0; i < 4; ++i) {
      unsigned off = threadIdx.x * 4 + i * 1024;
      unsigned lo = (unsigned int)(unsigned long long)(uintptr_t)&stg_v[off];
      unsigned long long gaS = (unsigned long long)(uintptr_t)(ov + dstb + base + off);
      asm volatile("global_store_async_from_lds_b128 %0, %1, off" :: "v"(gaS), "v"(lo) : "memory");
    }
    asm volatile("s_wait_asynccnt 0" ::: "memory");
  }
}

// =====================================================================
// Flash attention, WMMA bf16; epilogue writes ctx in swizzled A-layout.
// grid = (S/64, H), 128 threads = 4 waves, wave owns 16 query rows.
// =====================================================================
__global__ __launch_bounds__(128) void attn_kernel(
    const float* __restrict__ q,
    const float* __restrict__ kc,
    const float* __restrict__ vc,
    unsigned short* __restrict__ ctxs,   // swizzled A-layout [S x H*HD]
    const int* __restrict__ maskf)
{
  __shared__ v16bf KsF[2][2][32];
  __shared__ v16bf VsF[4][32];
  __shared__ v16bf PtF[4][32];

  const int tid  = threadIdx.x;
  const int lane = tid & 31;
  const int wave = tid >> 5;
  const int half = lane >> 4;
  const int lr   = lane & 15;
  const int klo  = half << 3;
  const int head = blockIdx.y;
  const int kv   = head >> 2;
  const int qblk = blockIdx.x * 64;
  const int qbase = qblk + wave * 16;
  const int flag = maskf[0];
  const float scale = 0.125f;

  unsigned short* ksp = (unsigned short*)KsF;
  unsigned short* vsp = (unsigned short*)VsF;
  unsigned short* ptp = (unsigned short*)PtF;

  v16bf aQ[2];
  {
    const float* qrow = q + (size_t)(qbase + lr) * (H_ * HD_) + head * HD_;
#pragma unroll
    for (int c = 0; c < 2; ++c) {
      const float* qp = qrow + c * 32 + klo;
      aQ[c] = frag16(*(const float4*)(qp), *(const float4*)(qp + 4),
                     *(const float4*)(qp + 16), *(const float4*)(qp + 20));
    }
  }

  v8f ctxa[4];
#pragma unroll
  for (int j = 0; j < 4; ++j) ctxa[j] = zero8();
  float mrun[8], lrun[8];
#pragma unroll
  for (int r = 0; r < 8; ++r) { mrun[r] = -1e30f; lrun[r] = 0.f; }

  const float* kchunk = kc + (size_t)kv * T_ * HD_;
  const float* vchunk = vc + (size_t)kv * T_ * HD_;
  const int keylimit = flag ? (qblk + 64) : T_;
  const int nch = (keylimit + 31) >> 5;
  const int wave_qmax = qbase + 15;

  const int vn = tid & 63;
  const int vk = (tid >> 6) * 4;

  for (int cix = 0; cix < nch; ++cix) {
    if (cix + 1 < nch) {
      __builtin_prefetch(kchunk + 32 * HD_ + tid * 16, 0, 1);
      __builtin_prefetch(vchunk + 32 * HD_ + tid * 16, 0, 1);
    }

#pragma unroll
    for (int it = 0; it < 4; ++it) {
      int off = (tid + it * 128) * 4;
      int key = off >> 6;
      int d   = off & 63;
      float4 fv = *(const float4*)(kchunk + (size_t)key * HD_ + d);
      int c     = d >> 5;
      int ki    = d & 31;
      int lslot = (((ki >> 3) & 1) << 4) + (key & 15);
      int e0    = (ki & 7) + ((ki >> 4) << 3);
      int t     = (key >> 4) & 1;
      pack4(&ksp[(((size_t)t * 2 + c) * 32 + lslot) * 16 + e0], fv.x, fv.y, fv.z, fv.w);
    }
#pragma unroll
    for (int p = 0; p < 4; ++p) {
      int key = p * 8 + vk;
      const float* vp = vchunk + (size_t)key * HD_ + vn;
      float x0 = vp[0];
      float x1 = vp[HD_];
      float x2 = vp[2 * HD_];
      float x3 = vp[3 * HD_];
      int jt    = vn >> 4;
      int lslot = (((key >> 3) & 1) << 4) + (vn & 15);
      int e0    = (key & 7) + ((key >> 4) << 3);
      pack4(&vsp[((size_t)jt * 32 + lslot) * 16 + e0], x0, x1, x2, x3);
    }
    __syncthreads();

    int j0 = cix * 32;
    if (!flag || j0 <= wave_qmax) {
      v8f s0 = zero8(), s1 = zero8();
#pragma unroll
      for (int c = 0; c < 2; ++c) {
        v16bf b0 = KsF[0][c][lane];
        v16bf b1 = KsF[1][c][lane];
        s0 = __builtin_amdgcn_wmma_f32_16x16x32_bf16(false, aQ[c], false, b0,
                                                     (short)0, s0, false, false);
        s1 = __builtin_amdgcn_wmma_f32_16x16x32_bf16(false, aQ[c], false, b1,
                                                     (short)0, s1, false, false);
      }

#pragma unroll
      for (int r = 0; r < 8; ++r) {
        float f0 = s0[r] * scale;
        float f1 = s1[r] * scale;
        int rowl = r + (half << 3);
        int rowi = qbase + rowl;
        if (flag) {
          if (j0 + lr      > rowi) f0 = -1e30f;
          if (j0 + 16 + lr > rowi) f1 = -1e30f;
        }
        float mc = fmaxf(f0, f1);
#pragma unroll
        for (int m = 8; m >= 1; m >>= 1) mc = fmaxf(mc, __shfl_xor(mc, m, 32));
        float nm = fmaxf(mrun[r], mc);
        float corr = __expf(mrun[r] - nm);
        float p0 = __expf(f0 - nm);
        float p1 = __expf(f1 - nm);
        float ls = p0 + p1;
#pragma unroll
        for (int m = 8; m >= 1; m >>= 1) ls += __shfl_xor(ls, m, 32);
        lrun[r] = lrun[r] * corr + ls;
        mrun[r] = nm;
#pragma unroll
        for (int j = 0; j < 4; ++j) ctxa[j][r] *= corr;
        int la = ((lr >> 3) & 1) * 16 + rowl;
        size_t pb = ((size_t)wave * 32 + la) * 16;
        ptp[pb + (lr & 7)]     = f2bf(p0);
        ptp[pb + (lr & 7) + 8] = f2bf(p1);
      }

      v16bf aP = PtF[wave][lane];
#pragma unroll
      for (int j = 0; j < 4; ++j) {
        v16bf bV = VsF[j][lane];
        ctxa[j] = __builtin_amdgcn_wmma_f32_16x16x32_bf16(false, aP, false, bV,
                                                          (short)0, ctxa[j],
                                                          false, false);
      }
    }
    __syncthreads();
    kchunk += 32 * HD_;
    vchunk += 32 * HD_;
  }

#pragma unroll
  for (int r = 0; r < 8; ++r) {
    float inv = 1.0f / lrun[r];
    int row = qbase + r + (half << 3);
#pragma unroll
    for (int j = 0; j < 4; ++j) {
      int kcol = head * HD_ + j * 16 + lr;
      ctxs[aswz_idx(row, kcol, H_ * HD_)] = f2bf(ctxa[j][r] * inv);
    }
  }
}

// =====================================================================
// SiLU(g)*u -> swizzled bf16 A-layout t
// =====================================================================
__global__ void silu_sw_kernel(const float* __restrict__ g,
                               const float* __restrict__ u,
                               unsigned short* __restrict__ ts)
{
  int idx = blockIdx.x * blockDim.x + threadIdx.x;
  if (idx >= S_ * (FF_ / 4)) return;
  int row = idx / (FF_ / 4);
  int c   = (idx - row * (FF_ / 4)) * 4;
  float4 gv = *(const float4*)(g + (size_t)row * FF_ + c);
  float4 uv = *(const float4*)(u + (size_t)row * FF_ + c);
  float y0 = (gv.x / (1.f + __expf(-gv.x))) * uv.x;
  float y1 = (gv.y / (1.f + __expf(-gv.y))) * uv.y;
  float y2 = (gv.z / (1.f + __expf(-gv.z))) * uv.z;
  float y3 = (gv.w / (1.f + __expf(-gv.w))) * uv.w;
  pack4(&ts[aswz_idx(row, c, FF_)], y0, y1, y2, y3);
}

// =====================================================================
// lm_head GEMV: logits[v] = sum_d hn[d] * W[d, v]
// =====================================================================
__global__ __launch_bounds__(256) void lmhead_kernel(
    const float* __restrict__ hn, const float* __restrict__ W,
    float* __restrict__ out)
{
  __shared__ float xs[D_];
  for (int i = threadIdx.x; i < D_; i += 256) xs[i] = hn[i];
  __syncthreads();
  int v = blockIdx.x * 256 + threadIdx.x;
  float acc = 0.f;
  for (int d = 0; d < D_; ++d) acc += xs[d] * W[(size_t)d * V_ + v];
  out[v] = acc;
}

// =====================================================================
// launcher
// =====================================================================
extern "C" void kernel_launch(void* const* d_in, const int* in_sizes, int n_in,
                              void* d_out, int out_size, void* d_ws, size_t ws_size,
                              hipStream_t stream)
{
  const float* hidden = (const float*)d_in[0];
  const float* past_k = (const float*)d_in[1];
  const float* past_v = (const float*)d_in[2];
  const float* ln1    = (const float*)d_in[3];
  const float* ln2    = (const float*)d_in[4];
  const float* wq     = (const float*)d_in[5];
  const float* wk     = (const float*)d_in[6];
  const float* wv     = (const float*)d_in[7];
  const float* wo     = (const float*)d_in[8];
  const float* wg     = (const float*)d_in[9];
  const float* wu     = (const float*)d_in[10];
  const float* wd     = (const float*)d_in[11];
  const float* norm_w = (const float*)d_in[12];
  const float* lmw    = (const float*)d_in[13];
  const float* cosT   = (const float*)d_in[14];
  const float* sinT   = (const float*)d_in[15];
  const int*   maskf  = (const int*)d_in[16];

  float* out    = (float*)d_out;
  float* logits = out;
  float* ok     = out + V_;
  float* ov     = ok + (size_t)L_ * KVH_ * T_ * HD_;

  float* ws = (float*)d_ws;
  float* h   = ws;                            // [S,D] fp32
  float* qb  = h  + (size_t)S_ * D_;          // [S,H*HD] fp32
  float* kb  = qb + (size_t)S_ * H_ * HD_;    // [S,KVH*HD] fp32
  float* vb  = kb + (size_t)S_ * KVH_ * HD_;
  float* g   = vb + (size_t)S_ * KVH_ * HD_;  // [S,FF] fp32
  float* u   = g  + (size_t)S_ * FF_;
  float* hl  = u  + (size_t)S_ * FF_;         // [D] fp32
  unsigned short* hn_sw  = (unsigned short*)(hl + 2048);          // [S,D] bf16 swz
  unsigned short* ctx_sw = hn_sw + (size_t)S_ * D_;               // [S,H*HD]
  unsigned short* t_sw   = ctx_sw + (size_t)S_ * H_ * HD_;        // [S,FF]
  unsigned short* bsw    = t_sw + (size_t)S_ * FF_;               // max K*N = D*FF

  (void)hipMemcpyAsync(h, hidden, sizeof(float) * S_ * D_, hipMemcpyDeviceToDevice, stream);

  past_copy_kernel<<<dim3(4, L_ * KVH_), 256, 0, stream>>>(past_k, past_v, ok, ov);

  for (int l = 0; l < L_; ++l) {
    const float* wq_l = wq + (size_t)l * D_ * H_ * HD_;
    const float* wk_l = wk + (size_t)l * D_ * KVH_ * HD_;
    const float* wv_l = wv + (size_t)l * D_ * KVH_ * HD_;
    const float* wo_l = wo + (size_t)l * H_ * HD_ * D_;
    const float* wg_l = wg + (size_t)l * D_ * FF_;
    const float* wu_l = wu + (size_t)l * D_ * FF_;
    const float* wd_l = wd + (size_t)l * FF_ * D_;
    float* okl = ok + (size_t)l * KVH_ * T_ * HD_;
    float* ovl = ov + (size_t)l * KVH_ * T_ * HD_;

    rmsnorm_sw_kernel<<<S_, 256, 0, stream>>>(h, ln1 + (size_t)l * D_, hn_sw, D_);

    wswz_kernel<<<(unsigned)(((size_t)D_ * H_ * HD_ / 4 + 255) / 256), 256, 0, stream>>>(wq_l, bsw, H_ * HD_, D_);
    gemm_sw<<<dim3(H_ * HD_ / BN, S_ / BM), 256, 0, stream>>>(hn_sw, bsw, qb, S_, H_ * HD_, D_, 0);
    wswz_kernel<<<(unsigned)(((size_t)D_ * KVH_ * HD_ / 4 + 255) / 256), 256, 0, stream>>>(wk_l, bsw, KVH_ * HD_, D_);
    gemm_sw<<<dim3(KVH_ * HD_ / BN, S_ / BM), 256, 0, stream>>>(hn_sw, bsw, kb, S_, KVH_ * HD_, D_, 0);
    wswz_kernel<<<(unsigned)(((size_t)D_ * KVH_ * HD_ / 4 + 255) / 256), 256, 0, stream>>>(wv_l, bsw, KVH_ * HD_, D_);
    gemm_sw<<<dim3(KVH_ * HD_ / BN, S_ / BM), 256, 0, stream>>>(hn_sw, bsw, vb, S_, KVH_ * HD_, D_, 0);

    rope_q_kernel<<<(S_ * H_ * (HD_ / 2) + 255) / 256, 256, 0, stream>>>(qb, cosT, sinT);
    kv_cache_kernel<<<(S_ * KVH_ * HD_ + 255) / 256, 256, 0, stream>>>(kb, vb, okl, ovl, cosT, sinT);

    attn_kernel<<<dim3(S_ / 64, H_), 128, 0, stream>>>(qb, okl, ovl, ctx_sw, maskf);

    wswz_kernel<<<(unsigned)(((size_t)H_ * HD_ * D_ / 4 + 255) / 256), 256, 0, stream>>>(wo_l, bsw, D_, H_ * HD_);
    gemm_sw<<<dim3(D_ / BN, S_ / BM), 256, 0, stream>>>(ctx_sw, bsw, h, S_, D_, H_ * HD_, 1);

    rmsnorm_sw_kernel<<<S_, 256, 0, stream>>>(h, ln2 + (size_t)l * D_, hn_sw, D_);

    wswz_kernel<<<(unsigned)(((size_t)D_ * FF_ / 4 + 255) / 256), 256, 0, stream>>>(wg_l, bsw, FF_, D_);
    gemm_sw<<<dim3(FF_ / BN, S_ / BM), 256, 0, stream>>>(hn_sw, bsw, g, S_, FF_, D_, 0);
    wswz_kernel<<<(unsigned)(((size_t)D_ * FF_ / 4 + 255) / 256), 256, 0, stream>>>(wu_l, bsw, FF_, D_);
    gemm_sw<<<dim3(FF_ / BN, S_ / BM), 256, 0, stream>>>(hn_sw, bsw, u, S_, FF_, D_, 0);

    silu_sw_kernel<<<(S_ * (FF_ / 4) + 255) / 256, 256, 0, stream>>>(g, u, t_sw);

    wswz_kernel<<<(unsigned)(((size_t)FF_ * D_ / 4 + 255) / 256), 256, 0, stream>>>(wd_l, bsw, D_, FF_);
    gemm_sw<<<dim3(D_ / BN, S_ / BM), 256, 0, stream>>>(t_sw, bsw, h, S_, D_, FF_, 1);
  }

  rmsnorm_kernel<<<1, 256, 0, stream>>>(h + (size_t)(S_ - 1) * D_, norm_w, hl, D_);
  lmhead_kernel<<<V_ / 256, 256, 0, stream>>>(hl, lmw, logits);
}
